// MatchingNoDecoder_66099546685447
// MI455X (gfx1250) — compile-verified, matrix operand order checked by reading, not run on hardware
//
#include <hip/hip_runtime.h>
#include <hip/hip_bf16.h>
#include <math.h>

// ---------------------------------------------------------------------------
// MI455X (gfx1250) pointer-network kernel suite.
// Big GEMMs use V_WMMA_F32_16X16X32_BF16 (one wave32 -> NT 16x16 f32 tiles).
// ---------------------------------------------------------------------------

typedef __attribute__((ext_vector_type(16))) __bf16 v16bf;
typedef __attribute__((ext_vector_type(8)))  float  v8f;
typedef unsigned int u32x4 __attribute__((ext_vector_type(4)));

#define BATCH   1024
#define NN      200     // n_nodes
#define NP      208     // n_nodes padded to 16
#define EMB     128
#define HID     256
#define G3      768     // 3*HID
#define KXI     224     // 200 padded to multiple of 32
#define NEGV    (-1000000000.0f)

// ---- WMMA bf16 fragment load, exact 16-bit A-matrix layout from ISA 7.12.2:
// lane%16 = matrix row; lane/16==0 holds K[k0..k0+7] and K[k0+16..k0+23],
// lane/16==1 holds K[k0+8..k0+15] and K[k0+24..k0+31].
union FragBF { v16bf v; u32x4 u[2]; };

__device__ __forceinline__ v16bf load_frag_bf16(const __bf16* __restrict__ base,
                                                int row, int ld, int k0, int half) {
  FragBF f;
  const __bf16* p = base + (size_t)row * ld + k0 + half * 8;
  f.u[0] = *(const u32x4*)(p);
  f.u[1] = *(const u32x4*)(p + 16);
  return f.v;
}

__device__ __forceinline__ v8f wmma_bf16(v16bf a, v16bf b, v8f c) {
  return __builtin_amdgcn_wmma_f32_16x16x32_bf16(false, a, false, b, (short)0, c,
                                                 false, false);
}

// ---------------------------------------------------------------------------
// 0) grid-stride zero fill (uint4 granularity)
// ---------------------------------------------------------------------------
__global__ void zero_kernel(u32x4* __restrict__ p, long n16) {
  long i = (long)blockIdx.x * blockDim.x + threadIdx.x;
  long stride = (long)gridDim.x * blockDim.x;
  u32x4 z = {0u, 0u, 0u, 0u};
  for (; i < n16; i += stride) p[i] = z;
}

// ---------------------------------------------------------------------------
// 1) f32 -> bf16 convert with zero padding (weights)
// ---------------------------------------------------------------------------
__global__ void convert_pad_kernel(const float* __restrict__ src, __bf16* __restrict__ dst,
                                   int rows_dst, int rows_src, int cols_dst, int cols_src) {
  int idx = blockIdx.x * blockDim.x + threadIdx.x;
  int total = rows_dst * cols_dst;
  if (idx >= total) return;
  int r = idx / cols_dst, c = idx % cols_dst;
  float v = (r < rows_src && c < cols_src) ? src[r * cols_src + c] : 0.0f;
  dst[idx] = (__bf16)v;
}

// ---------------------------------------------------------------------------
// 2) embedding: e{1,2}[b,n,:] = leaky_relu(x{1,2}[b,n,:2] @ W_emb^T + b_emb)
//    stored bf16, (BATCH, NP, EMB), pad rows (n>=200) = 0 exactly.
// ---------------------------------------------------------------------------
__global__ void embed_kernel(const float* __restrict__ x, const float* __restrict__ W_emb,
                             const float* __restrict__ b_emb,
                             __bf16* __restrict__ e1, __bf16* __restrict__ e2) {
  int idx = blockIdx.x * blockDim.x + threadIdx.x;
  const int total = BATCH * NP * EMB;
  if (idx >= total) return;
  int e = idx & (EMB - 1);
  int n = (idx / EMB) % NP;
  int b = idx / (EMB * NP);
  if (n >= NN) { e1[idx] = (__bf16)0.0f; e2[idx] = (__bf16)0.0f; return; }
  float w0 = W_emb[e * 2 + 0], w1 = W_emb[e * 2 + 1], bb = b_emb[e];
  const float* xb = x + (size_t)b * (2 * 2 * NN);
  float v1 = xb[n]      * w0 + xb[2 * NN + n]      * w1 + bb;
  float v2 = xb[NN + n] * w0 + xb[2 * NN + NN + n] * w1 + bb;
  v1 = v1 > 0.0f ? v1 : 0.01f * v1;
  v2 = v2 > 0.0f ? v2 : 0.01f * v2;
  e1[idx] = (__bf16)v1;
  e2[idx] = (__bf16)v2;
}

// ---------------------------------------------------------------------------
// 3) batched WMMA GEMM: xx[n,b,m] = sum_e e1[b,n,e]*e2[b,m,e]
//    xx buffer has NP (208) step-rows: rows 200..207 are dead padding so the
//    epilogue stores unconditionally (no divergent exec-mask chains).
// ---------------------------------------------------------------------------
__global__ void xx_gemm_kernel(const __bf16* __restrict__ e1, const __bf16* __restrict__ e2,
                               __bf16* __restrict__ xx) {
  const int TPB = NP / 16;                       // 13 tiles per dim
  int wave = (blockIdx.x * blockDim.x + threadIdx.x) >> 5;
  int total = BATCH * TPB * TPB;
  if (wave >= total) return;
  int b  = wave / (TPB * TPB);
  int t  = wave % (TPB * TPB);
  int tm = t / TPB, tn = t % TPB;
  int lane = threadIdx.x & 31;
  int half = lane >> 4, r = lane & 15;
  const __bf16* A = e1 + (size_t)b * NP * EMB;
  const __bf16* B = e2 + (size_t)b * NP * EMB;
  int arow = tm * 16 + r, brow = tn * 16 + r;
  v8f acc = {};
#pragma unroll
  for (int k = 0; k < EMB / 32; ++k) {
    v16bf af = load_frag_bf16(A, arow, EMB, k * 32, half);
    v16bf bf = load_frag_bf16(B, brow, EMB, k * 32, half);
    acc = wmma_bf16(af, bf, acc);
  }
  int m_g = tn * 16 + r;
#pragma unroll
  for (int i = 0; i < 8; ++i) {
    int n_g = tm * 16 + half * 8 + i;           // may be 200..207 -> pad rows
    xx[(size_t)n_g * BATCH * KXI + (size_t)b * KXI + m_g] = (__bf16)acc[i];
  }
}

// ---------------------------------------------------------------------------
// 4) WMMA GEMM, NT adjacent N-tiles per wave (A-fragment register reuse):
//    C(f32, MxN) = A(bf16, MxK rm) * B(bf16, NxK rm)^T ; K = 32*kIters.
// ---------------------------------------------------------------------------
template <int NT>
__global__ void gemm_bf16_nt_kernel(const __bf16* __restrict__ A, const __bf16* __restrict__ B,
                                    float* __restrict__ C, int tilesM, int tilesNgrp,
                                    int kIters, int lda, int ldb, int ldc) {
  int wave = (blockIdx.x * blockDim.x + threadIdx.x) >> 5;
  int total = tilesM * tilesNgrp;
  if (wave >= total) return;
  int tm = wave / tilesNgrp, tg = wave % tilesNgrp;
  int lane = threadIdx.x & 31;
  int half = lane >> 4, r = lane & 15;
  int arow = tm * 16 + r;
  const __bf16* ap = A + (size_t)arow * lda;
  v8f acc[NT];
#pragma unroll
  for (int j = 0; j < NT; ++j) acc[j] = (v8f){};
  for (int k = 0; k < kIters; ++k) {
    if (k + 1 < kIters) __builtin_prefetch(ap + (k + 1) * 32, 0, 0);
    v16bf af = load_frag_bf16(A, arow, lda, k * 32, half);
#pragma unroll
    for (int j = 0; j < NT; ++j) {
      int brow = (tg * NT + j) * 16 + r;
      v16bf bf = load_frag_bf16(B, brow, ldb, k * 32, half);
      acc[j] = wmma_bf16(af, bf, acc[j]);
    }
  }
  int rbase = tm * 16 + half * 8;
#pragma unroll
  for (int j = 0; j < NT; ++j) {
    int col = (tg * NT + j) * 16 + r;
#pragma unroll
    for (int i = 0; i < 8; ++i)
      C[(size_t)(rbase + i) * ldc + col] = acc[j][i];
  }
}

// ---------------------------------------------------------------------------
// 5) GRU pointwise gate combine; updates h (f32) and h_bf (bf16)
// ---------------------------------------------------------------------------
__global__ void gru_pointwise_kernel(const float* __restrict__ gi, const float* __restrict__ gh,
                                     const float* __restrict__ b_ih, const float* __restrict__ b_hh,
                                     float* __restrict__ h, __bf16* __restrict__ h_bf) {
  int idx = blockIdx.x * blockDim.x + threadIdx.x;
  if (idx >= BATCH * HID) return;
  int b = idx >> 8, j = idx & (HID - 1);
  const float* gib = gi + (size_t)b * G3;
  const float* ghb = gh + (size_t)b * G3;
  float ir  = gib[j]           + b_ih[j];
  float iz  = gib[HID + j]     + b_ih[HID + j];
  float in_ = gib[2 * HID + j] + b_ih[2 * HID + j];
  float hr  = ghb[j]           + b_hh[j];
  float hz  = ghb[HID + j]     + b_hh[HID + j];
  float hn  = ghb[2 * HID + j] + b_hh[2 * HID + j];
  float rg   = 1.0f / (1.0f + __expf(-(ir + hr)));
  float zg   = 1.0f / (1.0f + __expf(-(iz + hz)));
  float cand = tanhf(in_ + rg * hn);
  float ho = h[idx];
  float hnew = (1.0f - zg) * cand + zg * ho;
  h[idx] = hnew;
  h_bf[idx] = (__bf16)hnew;
}

// ---------------------------------------------------------------------------
// 6) per-step: logits -> mask -> log_softmax -> Gumbel-max sample (counter
//    hash; deterministic), write logps/gathered/actions/sels, update mask.
//    One wave32 per batch row.
// ---------------------------------------------------------------------------
__device__ __forceinline__ unsigned pcg_hash(unsigned v) {
  v = v * 747796405u + 2891336453u;
  unsigned w = ((v >> ((v >> 28u) + 4u)) ^ v) * 277803737u;
  return (w >> 22u) ^ w;
}

__global__ void sample_kernel(const float* __restrict__ logits, const float* __restrict__ b_out,
                              const float* __restrict__ x, unsigned char* __restrict__ mask,
                              float* __restrict__ logps, float* __restrict__ gathered,
                              float* __restrict__ actions, int* __restrict__ sels, int step) {
  int wave = (blockIdx.x * blockDim.x + threadIdx.x) >> 5;
  if (wave >= BATCH) return;
  int row = wave;
  int lane = threadIdx.x & 31;

  float lv[7];
#pragma unroll
  for (int i = 0; i < 7; ++i) {
    int m = lane + i * 32;
    float l = -3.0e38f;
    if (m < NN) {
      l = 10.0f * tanhf(logits[(size_t)row * NP + m] + b_out[m]);
      if (mask[(size_t)row * NP + m]) l = NEGV;
    }
    lv[i] = l;
  }

  // row max
  float mx = lv[0];
#pragma unroll
  for (int i = 1; i < 7; ++i) mx = fmaxf(mx, lv[i]);
  for (int off = 16; off > 0; off >>= 1) mx = fmaxf(mx, __shfl_xor(mx, off, 32));

  // log-sum-exp
  float s = 0.0f;
#pragma unroll
  for (int i = 0; i < 7; ++i) {
    int m = lane + i * 32;
    if (m < NN) s += __expf(lv[i] - mx);
  }
  for (int off = 16; off > 0; off >>= 1) s += __shfl_xor(s, off, 32);
  float lse = __logf(s);

  // write log-probs
  float* lp = logps + (size_t)step * BATCH * NN + (size_t)row * NN;
#pragma unroll
  for (int i = 0; i < 7; ++i) {
    int m = lane + i * 32;
    if (m < NN) lp[m] = lv[i] - mx - lse;
  }

  // Gumbel-max sampling (deterministic counter hash)
  float bestS = -3.0e38f, bestL = 0.0f;
  int bestI = 0;
#pragma unroll
  for (int i = 0; i < 7; ++i) {
    int m = lane + i * 32;
    if (m < NN) {
      unsigned hsh = pcg_hash((unsigned)(step * 1048576 + row * 256 + m) * 2654435761u + 0x9e3779b9u);
      float u = ((float)hsh + 0.5f) * (1.0f / 4294967296.0f);
      float g = -__logf(-__logf(u));
      float sc = lv[i] + g;
      if (sc > bestS) { bestS = sc; bestL = lv[i]; bestI = m; }
    }
  }
  for (int off = 16; off > 0; off >>= 1) {
    float os = __shfl_xor(bestS, off, 32);
    float ol = __shfl_xor(bestL, off, 32);
    int   oi = __shfl_xor(bestI, off, 32);
    if (os > bestS || (os == bestS && oi < bestI)) { bestS = os; bestL = ol; bestI = oi; }
  }

  if (lane == 0) {
    size_t so = (size_t)step * BATCH + row;
    sels[so] = bestI;
    gathered[so] = bestL - mx - lse;
    const float* xb = x + (size_t)row * (2 * 2 * NN);
    actions[so * 2 + 0] = xb[NN + bestI];             // x2[b,sel,0] = x[b,0,200+sel]
    actions[so * 2 + 1] = xb[2 * NN + NN + bestI];    // x2[b,sel,1] = x[b,1,200+sel]
    mask[(size_t)row * NP + bestI] = 1;
  }
}

// ---------------------------------------------------------------------------
// host launcher
// ---------------------------------------------------------------------------
extern "C" void kernel_launch(void* const* d_in, const int* in_sizes, int n_in,
                              void* d_out, int out_size, void* d_ws, size_t ws_size,
                              hipStream_t stream) {
  (void)in_sizes; (void)n_in; (void)out_size; (void)ws_size;

  const float* x     = (const float*)d_in[0];
  const float* W_emb = (const float*)d_in[1];
  const float* b_emb = (const float*)d_in[2];
  const float* W_ih  = (const float*)d_in[3];
  const float* b_ih  = (const float*)d_in[4];
  const float* W_hh  = (const float*)d_in[5];
  const float* b_hh  = (const float*)d_in[6];
  const float* W_out = (const float*)d_in[7];
  const float* b_out = (const float*)d_in[8];

  // ---- workspace carve-up (256B aligned) ----
  char* ws = (char*)d_ws;
  size_t off = 0;
  auto carve = [&](size_t bytes) -> char* {
    char* p = ws + off;
    off += (bytes + 255) & ~(size_t)255;
    return p;
  };
  __bf16* Wih_bf  = (__bf16*)carve((size_t)G3 * KXI * 2);
  __bf16* Whh_bf  = (__bf16*)carve((size_t)G3 * HID * 2);
  __bf16* Wout_bf = (__bf16*)carve((size_t)NP * HID * 2);
  float*  gi      = (float*)carve((size_t)BATCH * G3 * 4);
  float*  gh      = (float*)carve((size_t)BATCH * G3 * 4);
  float*  logitsb = (float*)carve((size_t)BATCH * NP * 4);
  size_t zero_begin = off;
  float*  h       = (float*)carve((size_t)BATCH * HID * 4);
  __bf16* h_bf    = (__bf16*)carve((size_t)BATCH * HID * 2);
  unsigned char* mask = (unsigned char*)carve((size_t)BATCH * NP);
  __bf16* xx      = (__bf16*)carve((size_t)NP * BATCH * KXI * 2);  // 208 step rows
  size_t zero_end = off;
  __bf16* e1      = (__bf16*)carve((size_t)BATCH * NP * EMB * 2);
  __bf16* e2      = (__bf16*)carve((size_t)BATCH * NP * EMB * 2);

  // ---- output carve-up ----
  float* out_f    = (float*)d_out;
  float* logps    = out_f;                                    // 200*1024*200
  float* gathered = logps + (size_t)NN * BATCH * NN;          // 200*1024
  float* actions  = gathered + (size_t)NN * BATCH;            // 200*1024*2
  int*   sels     = (int*)(actions + (size_t)NN * BATCH * 2); // 200*1024

  // 0) zero dynamic state (h, h_bf, mask, xx incl. K-pad columns)
  {
    long n16 = (long)((zero_end - zero_begin) / 16);
    zero_kernel<<<4096, 256, 0, stream>>>((u32x4*)(ws + zero_begin), n16);
  }

  // 1) weights -> bf16, zero padded
  {
    int t1 = G3 * KXI;
    convert_pad_kernel<<<(t1 + 255) / 256, 256, 0, stream>>>(W_ih, Wih_bf, G3, G3, KXI, NN);
    int t2 = G3 * HID;
    convert_pad_kernel<<<(t2 + 255) / 256, 256, 0, stream>>>(W_hh, Whh_bf, G3, G3, HID, HID);
    int t3 = NP * HID;
    convert_pad_kernel<<<(t3 + 255) / 256, 256, 0, stream>>>(W_out, Wout_bf, NP, NN, HID, HID);
  }

  // 2) embeddings (bf16, padded rows zero)
  {
    int total = BATCH * NP * EMB;
    embed_kernel<<<(total + 255) / 256, 256, 0, stream>>>(x, W_emb, b_emb, e1, e2);
  }

  // 3) xx = batched e1 @ e2^T   (WMMA bf16)
  {
    int waves = BATCH * (NP / 16) * (NP / 16);
    xx_gemm_kernel<<<(waves + 7) / 8, 256, 0, stream>>>(e1, e2, xx);
  }

  // 4) GRU scan, 200 sequential steps
  const int tilesM   = BATCH / 16;        // 64
  const int tilesNg4 = (G3 / 16) / 4;     // 12 groups of 4 N-tiles
  const int tilesNo  = NP / 16;           // 13
  for (int step = 0; step < NN; ++step) {
    const __bf16* xi = xx + (size_t)step * BATCH * KXI;
    {
      int waves = tilesM * tilesNg4;      // 768
      gemm_bf16_nt_kernel<4><<<(waves + 7) / 8, 256, 0, stream>>>(
          xi, Wih_bf, gi, tilesM, tilesNg4, KXI / 32, KXI, KXI, G3);
      gemm_bf16_nt_kernel<4><<<(waves + 7) / 8, 256, 0, stream>>>(
          h_bf, Whh_bf, gh, tilesM, tilesNg4, HID / 32, HID, HID, G3);
    }
    {
      int total = BATCH * HID;
      gru_pointwise_kernel<<<(total + 255) / 256, 256, 0, stream>>>(gi, gh, b_ih, b_hh, h, h_bf);
    }
    {
      int waves = tilesM * tilesNo;       // 832
      gemm_bf16_nt_kernel<1><<<(waves + 7) / 8, 256, 0, stream>>>(
          h_bf, Wout_bf, logitsb, tilesM, tilesNo, HID / 32, HID, HID, NP);
    }
    {
      int waves = BATCH;                  // one wave per row
      sample_kernel<<<(waves + 7) / 8, 256, 0, stream>>>(
          logitsb, b_out, x, mask, logps, gathered, actions, sels, step);
    }
  }
}
